// BiGRU_2886218023385
// MI455X (gfx1250) — matrix operand order repaired and checked by reading, not compiled, not run on hardware
//
#include <hip/hip_runtime.h>

// ---------------- problem constants ----------------
#define T_   256
#define B_   128
#define I_   128
#define H_   512
#define TBm  (T_ * B_)        // 32768 rows for the input-side GEMM
#define G3H  (3 * H_)         // 1536
#define H2   (2 * H_)         // 1024

typedef __attribute__((ext_vector_type(16))) __bf16 v16bf;
typedef __attribute__((ext_vector_type(8)))  float  v8f;
typedef __attribute__((__vector_size__(4 * sizeof(int)))) int v4i_t;

// ---------------- CDNA5 async global->LDS copy (ASYNCcnt path) -------------
#if defined(__has_builtin)
#  if __has_builtin(__builtin_amdgcn_global_load_async_to_lds_b128)
#    define USE_ASYNC_LDS 1
#  endif
#endif
#ifndef USE_ASYNC_LDS
#  define USE_ASYNC_LDS 0
#endif

// copy 16B (8 bf16) global -> LDS; async engine when available
static __device__ __forceinline__ void cp16(const __bf16* g, __bf16* l) {
#if USE_ASYNC_LDS
  __builtin_amdgcn_global_load_async_to_lds_b128(
      (__attribute__((address_space(1))) v4i_t*)(g),
      (__attribute__((address_space(3))) v4i_t*)(l), 0, 0);
#else
  *(uint4*)l = *(const uint4*)g;
#endif
}
template <int N>
static __device__ __forceinline__ void async_wait() {
#if USE_ASYNC_LDS
#  if __has_builtin(__builtin_amdgcn_s_wait_asynccnt)
  __builtin_amdgcn_s_wait_asynccnt(N);
#  else
  asm volatile("s_wait_asynccnt %0" ::"i"(N) : "memory");
#  endif
#endif
}

// ---------------- WMMA fragment loaders (CDNA5 wave32 16-bit layouts) ------
// A 16x32 (MxK): lane m = lane&15; VGPR v: K-pair = (v/4)*16 + (lane/16)*8 + (v%4)*2
static __device__ __forceinline__ v16bf ldfragA(const __bf16* base, int stride) {
  const int lane = threadIdx.x & 31;
  const __bf16* p = base + (lane & 15) * stride;
  const int h8 = (lane >> 4) * 8;
  v16bf f;
#pragma unroll
  for (int v = 0; v < 8; ++v) {
    int kb = ((v >> 2) << 4) + h8 + ((v & 3) << 1);
    f[2 * v]     = p[kb];
    f[2 * v + 1] = p[kb + 1];
  }
  return f;
}
// B 32x16 (KxN): lane n = lane&15; VGPR v: K-pair = (v/4)*16 + (lane/16)*8 + (v%4)*2
static __device__ __forceinline__ v16bf ldfragB(const __bf16* base, int stride) {
  const int lane = threadIdx.x & 31;
  const __bf16* p = base + (lane & 15) * stride;
  const int h8 = (lane >> 4) * 8;
  v16bf f;
#pragma unroll
  for (int v = 0; v < 8; ++v) {
    int kb = ((v >> 2) << 4) + h8 + ((v & 3) << 1);
    f[2 * v]     = p[kb];
    f[2 * v + 1] = p[kb + 1];
  }
  return f;
}

static __device__ __forceinline__ v8f zero_v8f() {
  v8f z;
#pragma unroll
  for (int i = 0; i < 8; ++i) z[i] = 0.0f;
  return z;
}
static __device__ __forceinline__ float sigmoidf(float x) {
  return 1.0f / (1.0f + __expf(-x));
}

// ---------------- fp32 -> bf16 conversion ----------------
__global__ void cvt_f32_bf16(const float* __restrict__ in, __bf16* __restrict__ out, size_t n) {
  size_t i = (size_t)blockIdx.x * blockDim.x + threadIdx.x;
  size_t stride = (size_t)gridDim.x * blockDim.x;
  for (; i < n; i += stride) out[i] = (__bf16)in[i];
}

__global__ void zero_u32(unsigned* p, int n) {
  if ((int)threadIdx.x < n) p[threadIdx.x] = 0u;
}

// ---------------- input-side GEMM: gx = X * Wih^T + bih --------------------
// Triple-buffered async pipeline: while tile i is consumed by WMMA, tile i+1
// streams into LDS on the async engine. 3 buffers -> single s_barrier per step.
#define ASTR 40   // padded LDS row stride (elements): 80B rows, b128-aligned, conflict-free
#define NBUF 3
__global__ __launch_bounds__(256)
void gx_gemm_kernel(const __bf16* __restrict__ X,     // [M,K]
                    const __bf16* __restrict__ Wih,   // [2][G3H][K]
                    const float*  __restrict__ bih,   // [2][G3H]
                    float*        __restrict__ gx,    // [2][M][G3H]
                    int K) {
  __shared__ __bf16 Alds[NBUF][128 * ASTR];   // 3 x 10KB
  __shared__ __bf16 Blds[NBUF][128 * ASTR];   // 3 x 10KB
  const int dir = blockIdx.z;
  const int n0  = blockIdx.x * 128;
  const int m0  = blockIdx.y * 128;
  const __bf16* Wd = Wih + (size_t)dir * G3H * K;
  const int tid = threadIdx.x;
  const int wv  = tid >> 5;
  const int row = tid >> 1, colh = (tid & 1) * 16;

  v8f acc[8];
#pragma unroll
  for (int c = 0; c < 8; ++c) acc[c] = zero_v8f();

  // issue one 128x32 A tile + B tile into LDS slot (4 x b128 per thread)
  auto issue = [&](int k0, int slot) {
    const __bf16* xs = X  + (size_t)(m0 + row) * K + k0 + colh;
    const __bf16* ws = Wd + (size_t)(n0 + row) * K + k0 + colh;
    cp16(xs,     &Alds[slot][row * ASTR + colh]);
    cp16(xs + 8, &Alds[slot][row * ASTR + colh + 8]);
    cp16(ws,     &Blds[slot][row * ASTR + colh]);
    cp16(ws + 8, &Blds[slot][row * ASTR + colh + 8]);
  };

  const int nk = K >> 5;
  issue(0, 0);
  for (int i = 0; i < nk; ++i) {
    if (i + 1 < nk) {
      issue((i + 1) << 5, (i + 1) % NBUF);
      async_wait<4>();   // previous tile's 4 transfers done; next 4 in flight
    } else {
      async_wait<0>();
    }
    __syncthreads();     // one barrier per K-step (3-buffer rotation)
    const int slot = i % NBUF;
    v16bf af = ldfragA(&Alds[slot][(wv * 16) * ASTR], ASTR);
#pragma unroll
    for (int c = 0; c < 8; ++c) {
      v16bf bf = ldfragB(&Blds[slot][(c * 16) * ASTR], ASTR);
      acc[c] = __builtin_amdgcn_wmma_f32_16x16x32_bf16(false, af, false, bf,
                                                       (short)0, acc[c], false, false);
    }
  }
  // epilogue: + bih, fp32 store. C-layout: lane n=lane&15, row m = v + 8*(lane/16)
  const int lane = tid & 31;
  const int mrow = m0 + wv * 16 + ((lane >> 4) << 3);
  const int ncb  = n0 + (lane & 15);
#pragma unroll
  for (int c = 0; c < 8; ++c) {
    int nc = ncb + c * 16;
    float bb = bih[dir * G3H + nc];
#pragma unroll
    for (int v = 0; v < 8; ++v)
      gx[((size_t)dir * TBm + mrow + v) * G3H + nc] = acc[c][v] + bb;
  }
}

// ---------------- persistent recurrent scan --------------------------------
// 32 WGs: dir = wg/16, 32-column j-slice of H. Resident LDS: this WG's 96 Whh
// rows (~100KB) + the full 128x512 bf16 h state re-staged each step (~133KB)
// => ~233KB of the 320KB WGP LDS. The 16-slice K loop runs with zero barriers;
// one async-wait + s_barrier + one device atomic barrier per timestep.
#define NWG  32
#define WSTR 520  // 1040B rows: 16B aligned, lane->bank stride 4 (conflict-free)
__device__ __forceinline__ void gbar(unsigned* c, unsigned tgt) {
  __syncthreads();
  if (threadIdx.x == 0) {
    __threadfence();
    __atomic_fetch_add(c, 1u, __ATOMIC_RELEASE);
    while (__atomic_load_n(c, __ATOMIC_ACQUIRE) < tgt)
      __builtin_amdgcn_s_sleep(2);
  }
  __syncthreads();
  __threadfence();
}

__global__ __launch_bounds__(256)
void scan_kernel(const float*  __restrict__ gx,    // [2][TBm][G3H]
                 const __bf16* __restrict__ Whh,   // [2][G3H][H]
                 const float*  __restrict__ bhh,   // [2][G3H]
                 float*        __restrict__ hF,    // [2][B][H]  f32 state
                 __bf16*       __restrict__ hB,    // [2][B][H]  bf16 state (WMMA A)
                 float*        __restrict__ outF,  // [T][B][2H]
                 __bf16*       __restrict__ outB,  // [T][B][2H]
                 unsigned*     __restrict__ cnt) {
  extern __shared__ __bf16 smem[];
  __bf16* Wlds  = smem;                 // [96][WSTR]  resident Whh slice
  __bf16* Afull = smem + 96 * WSTR;     // [128][WSTR] full h, re-staged per step
  const int tid = threadIdx.x, wv = tid >> 5, lane = tid & 31;
  const int dir = blockIdx.x >> 4;
  const int j0  = (blockIdx.x & 15) * 32;
  const __bf16* Wd = Whh + (size_t)dir * G3H * H_;

  // zero this WG's h-state slice
  for (int i = tid; i < 128 * 32; i += 256) {
    int b = i >> 5, j = j0 + (i & 31);
    hF[(dir * B_ + b) * H_ + j] = 0.0f;
    hB[(dir * B_ + b) * H_ + j] = (__bf16)0.0f;
  }
  // async-stage the 96 Whh rows {g*H + j0 + c} once (time-invariant)
  for (int idx = tid; idx < 96 * 64; idx += 256) {   // 16B chunks
    int r = idx >> 6, off = (idx & 63) * 8;
    int gr = (r >> 5) * H_ + j0 + (r & 31);
    cp16(Wd + (size_t)gr * H_ + off, &Wlds[r * WSTR + off]);
  }
  async_wait<0>();
  unsigned target = NWG;
  gbar(cnt, target);          // h zeroed everywhere + Wlds staged

  for (int s = 0; s < T_; ++s) {
    const int te = dir ? (T_ - 1 - s) : s;   // backward dir walks time reversed
    v8f acc[3][2];
#pragma unroll
    for (int g = 0; g < 3; ++g)
#pragma unroll
      for (int c = 0; c < 2; ++c) acc[g][c] = zero_v8f();

    // async-stage the FULL 128x512 h state into LDS (32 x b128 per thread)
    for (int idx = tid; idx < 128 * 64; idx += 256) {
      int r = idx >> 6, off = (idx & 63) * 8;
      cp16(hB + (size_t)(dir * B_ + r) * H_ + off, &Afull[r * WSTR + off]);
    }
    async_wait<0>();
    __syncthreads();

    // barrier-free K loop over all 16 slices
#pragma unroll 4
    for (int kk = 0; kk < H_; kk += 32) {
      v16bf af = ldfragA(&Afull[(wv * 16) * WSTR + kk], WSTR);
#pragma unroll
      for (int g = 0; g < 3; ++g)
#pragma unroll
        for (int c = 0; c < 2; ++c) {
          v16bf bf = ldfragB(&Wlds[(size_t)(g * 32 + c * 16) * WSTR + kk], WSTR);
          acc[g][c] = __builtin_amdgcn_wmma_f32_16x16x32_bf16(false, af, false, bf,
                                                              (short)0, acc[g][c], false, false);
        }
    }

    // gate math + h update (all three gates live in this WG's registers)
    const int bb  = wv * 16 + ((lane >> 4) << 3);
    const int jc0 = j0 + (lane & 15);
    const float* gxd = gx + ((size_t)dir * TBm + (size_t)te * B_) * G3H;
#pragma unroll
    for (int c = 0; c < 2; ++c) {
      int jc = jc0 + c * 16;
      float bhr = bhh[dir * G3H + jc];
      float bhz = bhh[dir * G3H + H_ + jc];
      float bhn = bhh[dir * G3H + 2 * H_ + jc];
#pragma unroll
      for (int v = 0; v < 8; ++v) {
        int b = bb + v;
        const float* gxb = gxd + (size_t)b * G3H;
        float r  = sigmoidf(gxb[jc]          + acc[0][c][v] + bhr);
        float z  = sigmoidf(gxb[H_ + jc]     + acc[1][c][v] + bhz);
        float n  = tanhf   (gxb[2 * H_ + jc] + r * (acc[2][c][v] + bhn));
        float hp = hF[(dir * B_ + b) * H_ + jc];
        float hn = (1.0f - z) * n + z * hp;
        hF[(dir * B_ + b) * H_ + jc] = hn;
        hB[(dir * B_ + b) * H_ + jc] = (__bf16)hn;
        size_t o = ((size_t)te * B_ + b) * H2 + dir * H_ + jc;
        outF[o] = hn;
        outB[o] = (__bf16)hn;
      }
    }
    target += NWG;
    gbar(cnt, target);   // publish h before any WG reads it next step
  }
}

// ---------------- final FC + sigmoid (wave-per-row dot product) ------------
__global__ __launch_bounds__(256)
void fc_kernel(const float* __restrict__ h,    // [TBm][2H]
               const float* __restrict__ w,    // [2H]
               const float* __restrict__ bias, // [1]
               float* __restrict__ out) {      // [TBm]
  int row  = blockIdx.x * 8 + (threadIdx.x >> 5);
  int lane = threadIdx.x & 31;
  const float* hr = h + (size_t)row * H2;
  float s = 0.0f;
  for (int k = lane; k < H2; k += 32) s += hr[k] * w[k];
#pragma unroll
  for (int off = 16; off; off >>= 1) s += __shfl_down(s, off, 32);
  if (lane == 0) out[row] = sigmoidf(s + bias[0]);
}

// ---------------- host orchestration ---------------------------------------
extern "C" void kernel_launch(void* const* d_in, const int* in_sizes, int n_in,
                              void* d_out, int out_size, void* d_ws, size_t ws_size,
                              hipStream_t stream) {
  const float* input_seq = (const float*)d_in[0];
  const float* W_ih0 = (const float*)d_in[1];
  const float* W_hh0 = (const float*)d_in[2];
  const float* b_ih0 = (const float*)d_in[3];
  const float* b_hh0 = (const float*)d_in[4];
  const float* W_ih  = (const float*)d_in[5];
  const float* W_hh  = (const float*)d_in[6];
  const float* b_ih  = (const float*)d_in[7];
  const float* b_hh  = (const float*)d_in[8];
  const float* fc_w  = (const float*)d_in[9];
  const float* fc_b  = (const float*)d_in[10];
  (void)in_sizes; (void)n_in; (void)out_size; (void)ws_size;

  // carve workspace
  char* p = (char*)d_ws;
  auto carve = [&](size_t bytes) { void* r = (void*)p; p += (bytes + 255) & ~(size_t)255; return r; };
  __bf16* xbf   = (__bf16*)carve((size_t)TBm * I_ * 2);
  __bf16* Wih0b = (__bf16*)carve((size_t)2 * G3H * I_ * 2);
  __bf16* Wihb  = (__bf16*)carve((size_t)4 * 2 * G3H * H2 * 2);
  __bf16* Whhb  = (__bf16*)carve((size_t)5 * 2 * G3H * H_ * 2);
  float*  gxbuf = (float*) carve((size_t)2 * TBm * G3H * 4);
  float*  oF0   = (float*) carve((size_t)TBm * H2 * 4);
  float*  oF1   = (float*) carve((size_t)TBm * H2 * 4);
  __bf16* oB0   = (__bf16*)carve((size_t)TBm * H2 * 2);
  __bf16* oB1   = (__bf16*)carve((size_t)TBm * H2 * 2);
  float*  hF    = (float*) carve((size_t)2 * B_ * H_ * 4);
  __bf16* hB    = (__bf16*)carve((size_t)2 * B_ * H_ * 2);
  unsigned* cnt = (unsigned*)carve(256);

  auto cvt = [&](const float* src, __bf16* dst, size_t n) {
    unsigned g = (unsigned)(((n + 255) / 256 < 4096) ? (n + 255) / 256 : 4096);
    cvt_f32_bf16<<<g, 256, 0, stream>>>(src, dst, n);
  };
  cvt(input_seq, xbf,  (size_t)TBm * I_);
  cvt(W_ih0,     Wih0b,(size_t)2 * G3H * I_);
  cvt(W_ih,      Wihb, (size_t)4 * 2 * G3H * H2);
  cvt(W_hh0,     Whhb, (size_t)2 * G3H * H_);
  cvt(W_hh,      Whhb + (size_t)2 * G3H * H_, (size_t)4 * 2 * G3H * H_);

  float*  oFs[2] = {oF0, oF1};
  __bf16* oBs[2] = {oB0, oB1};
  const size_t scan_smem = (size_t)(96 + 128) * WSTR * 2;  // ~233KB < 320KB/WGP
  const __bf16* X = xbf;
  int K = I_, pp = 0;
  for (int l = 0; l < 5; ++l) {
    const __bf16* Wi = (l == 0) ? Wih0b : Wihb + (size_t)(l - 1) * 2 * G3H * H2;
    const float*  bi = (l == 0) ? b_ih0 : b_ih + (size_t)(l - 1) * 2 * G3H;
    const __bf16* Wh = Whhb + (size_t)l * 2 * G3H * H_;
    const float*  bh = (l == 0) ? b_hh0 : b_hh + (size_t)(l - 1) * 2 * G3H;

    gx_gemm_kernel<<<dim3(G3H / 128, TBm / 128, 2), 256, 0, stream>>>(X, Wi, bi, gxbuf, K);
    zero_u32<<<1, 32, 0, stream>>>(cnt, 8);
    scan_kernel<<<NWG, 256, scan_smem, stream>>>(gxbuf, Wh, bh, hF, hB, oFs[pp], oBs[pp], cnt);

    X = oBs[pp];
    K = H2;
    pp ^= 1;
  }
  fc_kernel<<<TBm / 8, 256, 0, stream>>>(oFs[pp ^ 1], fc_w, fc_b, (float*)d_out);
}
